// S5Layer_10797547782359
// MI455X (gfx1250) — matrix operand order repaired and checked by reading, not compiled, not run on hardware
//
#include <hip/hip_runtime.h>

// ---------------------------------------------------------------------------
// S5 layer for MI455X (gfx1250, wave32, WMMA).
// Pipeline: matexp -> Bd -> [GEMM x*Dmat^T + x] -> [GEMM Bx] -> chunked scan
//           -> [GEMM += hs*C^T] -> in-place LayerNorm.
// Heavy GEMMs use bf16x3 split-precision WMMA (v_wmma_f32_16x16x32_bf16).
// LDS tiles are stored in WMMA-fragment order with an XOR chunk swizzle so
// each lane loads a whole v16bf operand with two conflict-free ds_load_b128.
// ---------------------------------------------------------------------------

typedef __attribute__((ext_vector_type(16))) __bf16 v16bf;
typedef __attribute__((ext_vector_type(8)))  __bf16 v8bf;
typedef __attribute__((ext_vector_type(4)))  __bf16 v4bf;
typedef __attribute__((ext_vector_type(8)))  float  v8f;

#define D_MODEL 1024
#define D_STATE 64
#define BATCH   8
#define SEQ     4096
#define NROWS   (BATCH * SEQ)          // 32768
#define CHUNK   64                     // scan chunk length
#define NCHUNK  (SEQ / CHUNK)          // 64
#define LN_EPS  1e-5f

__device__ __forceinline__ __bf16 f2bf(float f) { return (__bf16)f; }
__device__ __forceinline__ float  bf2f(__bf16 h) { return (float)h; }

// ---------------------------------------------------------------------------
// Kernel 0: one workgroup. Computes
//   Ad  = expm(A*dt)          (scaling-and-squaring, s=6, Taylor-8)
//   Pdt = dt * phi(A*dt)      (phi(Z) = Z^-1(e^Z - I); doubling identity)
//   AdL = Ad^CHUNK            (for the chunked scan carry)
// ---------------------------------------------------------------------------
__global__ __launch_bounds__(256) void matexp_kernel(
    const float* __restrict__ A, const float* __restrict__ log_dt,
    float* __restrict__ Ad, float* __restrict__ AdL, float* __restrict__ Pdt)
{
    __shared__ float b0[4096];   // X during Taylor; scratch afterwards
    __shared__ float b1[4096];   // Cterm, then E
    __shared__ float b2[4096];   // P
    const int t  = threadIdx.x;
    const int r  = t >> 2;             // output row (0..63)
    const int c0 = (t & 3) * 16;       // output col base (16 cols per thread)
    const float dt = expf(log_dt[0]);

    // X = A*dt/64 ;  Cterm = I
    for (int i = 0; i < 16; ++i) {
        int idx = t * 16 + i;
        b0[idx] = A[idx] * dt * (1.0f / 64.0f);
        int rr = idx >> 6, cc = idx & 63;
        b1[idx] = (rr == cc) ? 1.0f : 0.0f;
    }
    float Ereg[16], Preg[16];
    #pragma unroll
    for (int i = 0; i < 16; ++i) {
        float id = (r == (c0 + i)) ? 1.0f : 0.0f;
        Ereg[i] = id; Preg[i] = id;           // k=0 terms
    }
    __syncthreads();

    // Taylor: C_k = C_{k-1} * X / k ; E += C_k ; P += C_k/(k+1)
    for (int k = 1; k <= 8; ++k) {
        float tmp[16];
        #pragma unroll
        for (int i = 0; i < 16; ++i) {
            float s = 0.0f;
            for (int j = 0; j < 64; ++j) s += b1[r * 64 + j] * b0[j * 64 + c0 + i];
            tmp[i] = s / (float)k;
        }
        __syncthreads();
        #pragma unroll
        for (int i = 0; i < 16; ++i) {
            b1[r * 64 + c0 + i] = tmp[i];
            Ereg[i] += tmp[i];
            Preg[i] += tmp[i] / (float)(k + 1);
        }
        __syncthreads();
    }

    // b1 = E, b2 = P
    #pragma unroll
    for (int i = 0; i < 16; ++i) { b1[r*64 + c0 + i] = Ereg[i]; b2[r*64 + c0 + i] = Preg[i]; }
    __syncthreads();

    // 6 doublings: P' = (E+I)P/2 ; E' = E*E
    for (int sq = 0; sq < 6; ++sq) {
        float en[16], pn[16];
        #pragma unroll
        for (int i = 0; i < 16; ++i) {
            float se = 0.0f, sp = 0.0f;
            for (int j = 0; j < 64; ++j) {
                float erj = b1[r * 64 + j];
                se += erj * b1[j * 64 + c0 + i];
                sp += erj * b2[j * 64 + c0 + i];
            }
            en[i] = se;
            pn[i] = 0.5f * (sp + b2[r * 64 + c0 + i]);
        }
        __syncthreads();
        #pragma unroll
        for (int i = 0; i < 16; ++i) { b1[r*64 + c0 + i] = en[i]; b2[r*64 + c0 + i] = pn[i]; }
        __syncthreads();
    }

    for (int i = 0; i < 16; ++i) {
        int idx = t * 16 + i;
        Ad[idx]  = b1[idx];
        Pdt[idx] = dt * b2[idx];
    }
    __syncthreads();

    // 6 more squarings: AdL = Ad^64
    for (int sq = 0; sq < 6; ++sq) {
        float en[16];
        #pragma unroll
        for (int i = 0; i < 16; ++i) {
            float se = 0.0f;
            for (int j = 0; j < 64; ++j) se += b1[r * 64 + j] * b1[j * 64 + c0 + i];
            en[i] = se;
        }
        __syncthreads();
        #pragma unroll
        for (int i = 0; i < 16; ++i) b1[r * 64 + c0 + i] = en[i];
        __syncthreads();
    }
    for (int i = 0; i < 16; ++i) AdL[t * 16 + i] = b1[t * 16 + i];
}

// ---------------------------------------------------------------------------
// Kernel: Bd[n][d] = sum_k Pdt[n][k] * B[k][d]   (64x1024)
// ---------------------------------------------------------------------------
__global__ __launch_bounds__(256) void bd_kernel(
    const float* __restrict__ Pdt, const float* __restrict__ B, float* __restrict__ Bd)
{
    int idx = blockIdx.x * 256 + threadIdx.x;     // 0..65535
    int n = idx >> 10, d = idx & 1023;
    float s = 0.0f;
    #pragma unroll 8
    for (int k = 0; k < 64; ++k) s += Pdt[n * 64 + k] * B[k * 1024 + d];
    Bd[idx] = s;
}

// ---------------------------------------------------------------------------
// bf16x3 split-precision WMMA GEMM:
//   Cout[m][n] (+)= sum_k A[m][k] * W[n][k]
// A: MxK row-major f32, W: NxK row-major f32 (i.e. B-operand transposed).
// Block: 256 threads = 8 waves (4M x 2N); wave tile 32x32 (2x2 of 16x16).
// Block tile 128(M) x 64(N), K-step 32.
//
// LDS layout = WMMA fragment order. Per 16x32 (A) / 32x16 (B) subtile a
// region of 512 halfs holds, for each lane, two XOR-swizzled 16B chunks:
//   chunk(lane, g) = lane*2 + (g ^ ((lane>>3)&1)),   g = j>>3
// so a lane's v16bf is two ds_load_b128 hitting each 16B bank-group exactly
// twice across the wave (conflict-free floor).
//
// Fragment element maps (from the CDNA5 16-bit layouts):
//   A: lane = 8*((k>>3)&1)*2 + (r&15) ... K = (g<<4)|(half<<3)|(j&7)
//   B: lane = 16*((k>>3)&1) + (n&15)  ... K = (g<<4)|(half<<3)|(j&7)
// mode: 0 = store, 1 = store + addend[m][n], 2 = accumulate into Cout.
// ---------------------------------------------------------------------------
#define GTM 128
#define GTN 64
#define GTK 32

// Staging address (in halfs) for A-tile element (row r in 0..127, k in 0..31).
__device__ __forceinline__ int fragA_off(int r, int k)
{
    int m_sub = r >> 4, l16 = r & 15;
    int half  = (k >> 3) & 1;
    int lane  = half * 16 + l16;
    int g     = (k >> 4) & 1;
    int sw    = (lane >> 3) & 1;
    return m_sub * 512 + (lane * 2 + (g ^ sw)) * 8 + (k & 7);
}

// Staging address (in halfs) for W-tile element (row n in 0..63, k in 0..31).
__device__ __forceinline__ int fragB_off(int n, int k)
{
    int n_sub = n >> 4, l16 = n & 15;
    int half  = (k >> 3) & 1;
    int lane  = half * 16 + l16;
    int g     = (k >> 4) & 1;
    int sw    = (lane >> 3) & 1;
    return n_sub * 512 + (lane * 2 + (g ^ sw)) * 8 + (k & 7);
}

// Load one v16bf fragment: two swizzled ds_load_b128 from a subtile region.
__device__ __forceinline__ v16bf load_frag(const __bf16* base, int lane)
{
    int sw = (lane >> 3) & 1;
    v8bf f0 = *(const v8bf*)(base + (lane * 2 + sw) * 8);        // g = 0
    v8bf f1 = *(const v8bf*)(base + (lane * 2 + (1 - sw)) * 8);  // g = 1
    return __builtin_shufflevector(f0, f1, 0, 1, 2, 3, 4, 5, 6, 7,
                                           8, 9, 10, 11, 12, 13, 14, 15);
}

__global__ __launch_bounds__(256) void gemm_bf16x3(
    const float* __restrict__ A, int lda,
    const float* __restrict__ W, int ldw,
    float* __restrict__ Cout, int ldc,
    const float* __restrict__ addend,
    int K, int mode)
{
    __shared__ __bf16 sa_hi[8 * 512];   // 8 M-subtiles, fragment order
    __shared__ __bf16 sa_lo[8 * 512];
    __shared__ __bf16 sw_hi[4 * 512];   // 4 N-subtiles, fragment order
    __shared__ __bf16 sw_lo[4 * 512];

    const int tid  = threadIdx.x;
    const int lane = tid & 31;
    const int wave = tid >> 5;
    const int wm   = (wave >> 1) * 32;     // wave M offset within block tile
    const int wn   = (wave & 1)  * 32;     // wave N offset within block tile
    const int half = lane >> 4;            // lane group (0: lanes 0-15, 1: 16-31)
    const int l16  = lane & 15;
    const int m0   = blockIdx.x * GTM;
    const int n0   = blockIdx.y * GTN;

    v8f acc[2][2] = {};

    for (int k0 = 0; k0 < K; k0 += GTK) {
        // ---- stage A tile: 128x32 f32 -> bf16 hi/lo (1024 float4, 4/thread)
        #pragma unroll
        for (int i = 0; i < 4; ++i) {
            int idx = tid + i * 256;
            int r = idx >> 3, c = (idx & 7) * 4;     // c = 0,4,...,28
            const float4 v = *(const float4*)(A + (size_t)(m0 + r) * lda + k0 + c);
            float f[4] = {v.x, v.y, v.z, v.w};
            v4bf vh, vl;
            #pragma unroll
            for (int j = 0; j < 4; ++j) {
                __bf16 h = f2bf(f[j]);
                vh[j] = h;
                vl[j] = f2bf(f[j] - bf2f(h));
            }
            int off = fragA_off(r, c);               // 4 halfs share one chunk
            *(v4bf*)&sa_hi[off] = vh;                // ds_store_b64
            *(v4bf*)&sa_lo[off] = vl;
            if (k0 + GTK < K)  // warm next K-tile (global_prefetch_b8)
                __builtin_prefetch(A + (size_t)(m0 + r) * lda + k0 + GTK + c, 0, 1);
        }
        // ---- stage W tile: 64x32 f32 -> bf16 hi/lo (512 float4, 2/thread)
        #pragma unroll
        for (int i = 0; i < 2; ++i) {
            int idx = tid + i * 256;
            int rn = idx >> 3, c = (idx & 7) * 4;
            const float4 v = *(const float4*)(W + (size_t)(n0 + rn) * ldw + k0 + c);
            float f[4] = {v.x, v.y, v.z, v.w};
            v4bf vh, vl;
            #pragma unroll
            for (int j = 0; j < 4; ++j) {
                __bf16 h = f2bf(f[j]);
                vh[j] = h;
                vl[j] = f2bf(f[j] - bf2f(h));
            }
            int off = fragB_off(rn, c);
            *(v4bf*)&sw_hi[off] = vh;
            *(v4bf*)&sw_lo[off] = vl;
        }
        __syncthreads();

        // ---- fragment loads: 2 x ds_load_b128 per v16bf operand
        v16bf ah[2], al[2], bh[2], bl[2];
        #pragma unroll
        for (int mi = 0; mi < 2; ++mi) {
            const int msub = (wm >> 4) + mi;
            ah[mi] = load_frag(&sa_hi[msub * 512], lane);
            al[mi] = load_frag(&sa_lo[msub * 512], lane);
        }
        #pragma unroll
        for (int ni = 0; ni < 2; ++ni) {
            const int nsub = (wn >> 4) + ni;
            bh[ni] = load_frag(&sw_hi[nsub * 512], lane);
            bl[ni] = load_frag(&sw_lo[nsub * 512], lane);
        }

        // ---- 3 WMMAs per subtile: hi*hi + hi*lo + lo*hi  (~fp32 accuracy)
        #pragma unroll
        for (int mi = 0; mi < 2; ++mi)
            #pragma unroll
            for (int ni = 0; ni < 2; ++ni) {
                acc[mi][ni] = __builtin_amdgcn_wmma_f32_16x16x32_bf16(
                    false, ah[mi], false, bh[ni], (short)0, acc[mi][ni], false, false);
                acc[mi][ni] = __builtin_amdgcn_wmma_f32_16x16x32_bf16(
                    false, ah[mi], false, bl[ni], (short)0, acc[mi][ni], false, false);
                acc[mi][ni] = __builtin_amdgcn_wmma_f32_16x16x32_bf16(
                    false, al[mi], false, bh[ni], (short)0, acc[mi][ni], false, false);
            }
        __syncthreads();
    }

    // ---- epilogue (C/D layout: elem r -> M = r + 8*half, N = lane%16)
    #pragma unroll
    for (int mi = 0; mi < 2; ++mi)
        #pragma unroll
        for (int ni = 0; ni < 2; ++ni)
            #pragma unroll
            for (int r = 0; r < 8; ++r) {
                int row = m0 + wm + mi * 16 + r + 8 * half;
                int col = n0 + wn + ni * 16 + l16;
                size_t idx = (size_t)row * ldc + col;
                float v = acc[mi][ni][r];
                if (mode == 1) v += addend[idx];
                if (mode == 2) v += Cout[idx];
                Cout[idx] = v;
            }
}

// ---------------------------------------------------------------------------
// Scan pass A: per-chunk local recurrence with h0=0; emit chunk summary v_c.
// 512 threads: thread t owns (b = t>>6, n = t&63). A_d^T column in registers.
// ---------------------------------------------------------------------------
__global__ __launch_bounds__(512) void scan_local(
    const float* __restrict__ Bx, const float* __restrict__ Ad, float* __restrict__ chunkv)
{
    __shared__ float adt[64][64];   // adt[k][n] = Ad[n][k] (conflict-free reads)
    __shared__ float hbuf[8][64];
    const int t = threadIdx.x, b = t >> 6, n = t & 63;
    for (int i = 0; i < 8; ++i) {
        int idx = t + i * 512; int rn = idx >> 6, k = idx & 63;
        adt[k][rn] = Ad[rn * 64 + k];
    }
    hbuf[b][n] = 0.0f;
    __syncthreads();
    float acol[64];
    #pragma unroll
    for (int k = 0; k < 64; ++k) acol[k] = adt[k][n];

    const int c = blockIdx.x;
    const float* bx = Bx + ((size_t)b * SEQ + c * CHUNK) * 64 + n;
    for (int s = 0; s < CHUNK; ++s) {
        float sum = bx[s * 64];
        #pragma unroll
        for (int k = 0; k < 64; ++k) sum += hbuf[b][k] * acol[k];
        __syncthreads();
        hbuf[b][n] = sum;
        __syncthreads();
    }
    chunkv[(size_t)c * 512 + t] = hbuf[b][n];
}

// ---------------------------------------------------------------------------
// Scan pass B: single-block sequential carry over 64 chunk summaries using
// P = (A_d^T)^CHUNK. Emits per-chunk start states.
// ---------------------------------------------------------------------------
__global__ __launch_bounds__(512) void scan_carry(
    const float* __restrict__ chunkv, const float* __restrict__ AdL, float* __restrict__ hstart)
{
    __shared__ float adt[64][64];
    __shared__ float hbuf[8][64];
    const int t = threadIdx.x, b = t >> 6, n = t & 63;
    for (int i = 0; i < 8; ++i) {
        int idx = t + i * 512; int rn = idx >> 6, k = idx & 63;
        adt[k][rn] = AdL[rn * 64 + k];
    }
    hbuf[b][n] = 0.0f;
    __syncthreads();
    float acol[64];
    #pragma unroll
    for (int k = 0; k < 64; ++k) acol[k] = adt[k][n];

    for (int c = 0; c < NCHUNK; ++c) {
        hstart[(size_t)c * 512 + t] = hbuf[b][n];
        float sum = chunkv[(size_t)c * 512 + t];
        #pragma unroll
        for (int k = 0; k < 64; ++k) sum += hbuf[b][k] * acol[k];
        __syncthreads();
        hbuf[b][n] = sum;
        __syncthreads();
    }
}

// ---------------------------------------------------------------------------
// Scan pass C: replay each chunk seeded with corrected start state; write hs.
// ---------------------------------------------------------------------------
__global__ __launch_bounds__(512) void scan_apply(
    const float* __restrict__ Bx, const float* __restrict__ Ad,
    const float* __restrict__ hstart, float* __restrict__ HS)
{
    __shared__ float adt[64][64];
    __shared__ float hbuf[8][64];
    const int t = threadIdx.x, b = t >> 6, n = t & 63;
    for (int i = 0; i < 8; ++i) {
        int idx = t + i * 512; int rn = idx >> 6, k = idx & 63;
        adt[k][rn] = Ad[rn * 64 + k];
    }
    const int c = blockIdx.x;
    hbuf[b][n] = hstart[(size_t)c * 512 + t];
    __syncthreads();
    float acol[64];
    #pragma unroll
    for (int k = 0; k < 64; ++k) acol[k] = adt[k][n];

    const float* bx = Bx + ((size_t)b * SEQ + c * CHUNK) * 64 + n;
    for (int s = 0; s < CHUNK; ++s) {
        float sum = bx[s * 64];
        #pragma unroll
        for (int k = 0; k < 64; ++k) sum += hbuf[b][k] * acol[k];
        __syncthreads();
        hbuf[b][n] = sum;
        HS[((size_t)b * SEQ + c * CHUNK + s) * 64 + n] = sum;
        __syncthreads();
    }
}

// ---------------------------------------------------------------------------
// In-place LayerNorm over rows of 1024. One block (256 threads) per row.
// ---------------------------------------------------------------------------
__global__ __launch_bounds__(256) void ln_kernel(
    float* __restrict__ y, const float* __restrict__ gamma, const float* __restrict__ beta)
{
    __shared__ float red[256];
    const int t = threadIdx.x;
    float* p = y + (size_t)blockIdx.x * D_MODEL;
    float v[4]; float s = 0.0f;
    #pragma unroll
    for (int i = 0; i < 4; ++i) { v[i] = p[t + i * 256]; s += v[i]; }
    red[t] = s; __syncthreads();
    for (int o = 128; o > 0; o >>= 1) { if (t < o) red[t] += red[t + o]; __syncthreads(); }
    float mu = red[0] * (1.0f / D_MODEL);
    __syncthreads();
    s = 0.0f;
    #pragma unroll
    for (int i = 0; i < 4; ++i) { float d = v[i] - mu; s += d * d; }
    red[t] = s; __syncthreads();
    for (int o = 128; o > 0; o >>= 1) { if (t < o) red[t] += red[t + o]; __syncthreads(); }
    float rs = rsqrtf(red[0] * (1.0f / D_MODEL) + LN_EPS);
    #pragma unroll
    for (int i = 0; i < 4; ++i) {
        int col = t + i * 256;
        p[col] = (v[i] - mu) * rs * gamma[col] + beta[col];
    }
}

// ---------------------------------------------------------------------------
extern "C" void kernel_launch(void* const* d_in, const int* in_sizes, int n_in,
                              void* d_out, int out_size, void* d_ws, size_t ws_size,
                              hipStream_t stream) {
    const float* x      = (const float*)d_in[0];   // [8,4096,1024]
    const float* A      = (const float*)d_in[1];   // [64,64]
    const float* B      = (const float*)d_in[2];   // [64,1024]
    const float* Cmat   = (const float*)d_in[3];   // [1024,64]
    const float* Dmat   = (const float*)d_in[4];   // [1024,1024]
    const float* log_dt = (const float*)d_in[5];   // [1]
    const float* gamma  = (const float*)d_in[6];   // [1024]
    const float* beta   = (const float*)d_in[7];   // [1024]
    float* out = (float*)d_out;                    // [8,4096,1024]

    // workspace layout (floats); total ~17.4 MB
    float* ws     = (float*)d_ws;
    float* Bx     = ws;                      // NROWS*64      = 2097152
    float* HS     = Bx  + 2097152;           // NROWS*64      = 2097152
    float* Ad     = HS  + 2097152;           // 4096
    float* AdL    = Ad  + 4096;              // 4096
    float* Pdt    = AdL + 4096;              // 4096
    float* Bd     = Pdt + 4096;              // 64*1024       = 65536
    float* chunkv = Bd  + 65536;             // 64*512        = 32768
    float* hstart = chunkv + 32768;          // 64*512        = 32768

    // 0) discretization: Ad = expm(A dt), Pdt = dt*phi(A dt), AdL = Ad^64
    matexp_kernel<<<1, 256, 0, stream>>>(A, log_dt, Ad, AdL, Pdt);
    // 0b) Bd = Pdt @ B
    bd_kernel<<<256, 256, 0, stream>>>(Pdt, B, Bd);

    // 1) big GEMM: out = x @ Dmat^T + x   (residual fused into epilogue)
    gemm_bf16x3<<<dim3(NROWS / GTM, D_MODEL / GTN), 256, 0, stream>>>(
        x, D_MODEL, Dmat, D_MODEL, out, D_MODEL, x, D_MODEL, 1);

    // 2) Bx = x @ Bd^T   ([NROWS,64])
    gemm_bf16x3<<<dim3(NROWS / GTM, D_STATE / GTN), 256, 0, stream>>>(
        x, D_MODEL, Bd, D_MODEL, Bx, D_STATE, nullptr, D_MODEL, 0);

    // 3) chunked linear scan
    scan_local<<<NCHUNK, 512, 0, stream>>>(Bx, Ad, chunkv);
    scan_carry<<<1, 512, 0, stream>>>(chunkv, AdL, hstart);
    scan_apply<<<NCHUNK, 512, 0, stream>>>(Bx, Ad, hstart, HS);

    // 4) out += hs @ C^T
    gemm_bf16x3<<<dim3(NROWS / GTM, D_MODEL / GTN), 256, 0, stream>>>(
        HS, D_STATE, Cmat, D_STATE, out, D_MODEL, nullptr, D_STATE, 2);

    // 5) in-place LayerNorm
    ln_kernel<<<NROWS, 256, 0, stream>>>(out, gamma, beta);
}
